// QuantLinear_82025285419484
// MI455X (gfx1250) — matrix operand order repaired and checked by reading, not compiled, not run on hardware
//
#include <hip/hip_runtime.h>

#define IN_F   4096
#define OUT_F  4096
#define TOKENS 8192
#define LR     16
#define SCALING 2.0f
#define KSTEPS (IN_F / 32)

typedef __attribute__((ext_vector_type(16))) _Float16 v16h;
typedef __attribute__((ext_vector_type(2)))  _Float16 h2;
typedef __attribute__((ext_vector_type(8)))  float    v8f;
typedef __attribute__((ext_vector_type(4)))  unsigned u32x4;
typedef __attribute__((ext_vector_type(4)))  float    f32x4;

struct U8x32 { u32x4 lo, hi; };

// packed f32 -> 2 x f16 (v_cvt_pk_rtz_f16_f32), returned as bits
static __device__ __forceinline__ unsigned packh(float a, float b) {
  return __builtin_bit_cast(unsigned, __builtin_amdgcn_cvt_pkrtz(a, b));
}
static __device__ __forceinline__ v16h mkfrag(u32x4 lo, u32x4 hi) {
  U8x32 u{lo, hi};
  return __builtin_bit_cast(v16h, u);
}

// K-permutation note: within every 8-element K-chunk, position p holds original
// k = (p>>1) + 4*(p&1). Applied identically to xbf, lora_A frags, and the
// dequant output, so the GEMM sum over K is unchanged but the dequant needs
// zero reordering instructions (nibble i and i+4 come out of one v_and_or).

// ---------------------------------------------------------------------------
// Kernel 0: xbf = f16(x), K-swizzled. One-time pass so the 32 N-tile re-reads
// of x move f16 through L2 with no per-tile conversion. 8 elems / thread.
// ---------------------------------------------------------------------------
__global__ void __launch_bounds__(256)
x_to_f16_kernel(const float* __restrict__ x, unsigned short* __restrict__ xbf) {
  const size_t i = ((size_t)blockIdx.x * 256 + threadIdx.x) * 8;
  const f32x4* p = reinterpret_cast<const f32x4*>(x + i);
  f32x4 v0 = p[0], v1 = p[1];                       // k = 0..3, 4..7 of chunk
  u32x4 o = { packh(v0.x, v1.x), packh(v0.y, v1.y),  // swizzle: (0,4)(1,5)(2,6)(3,7)
              packh(v0.z, v1.z), packh(v0.w, v1.w) };
  *reinterpret_cast<u32x4*>(xbf + i) = o;
}

// ---------------------------------------------------------------------------
// Kernel 1: T[8192,16] = x @ lora_A^T via WMMA (A-frags straight from xbf).
// One block = 128 token rows = 8 waves, each wave one 16x16 (M x R) tile.
// ---------------------------------------------------------------------------
__global__ void __launch_bounds__(256)
lora_t_kernel(const unsigned short* __restrict__ xbf, const float* __restrict__ A,
              float* __restrict__ T) {
  const int t    = threadIdx.x;
  const int wave = t >> 5, lane = t & 31;
  const int lm   = lane & 15;          // M (A-frag) / N==r (B-frag)
  const int hk   = lane >> 4;          // lane-group selects K half
  const int m0   = blockIdx.x * 128;
  const int row  = m0 + wave * 16 + lm;

  v8f acc = {};
  for (int k0 = 0; k0 < IN_F; k0 += 32) {
    // A-frag: f16 rows (already swizzled), K-pos = kb..kb+15 (16 contiguous per lane group)
    const unsigned short* ap = xbf + (size_t)row * IN_F + k0 + hk * 16;
    u32x4 alo = *reinterpret_cast<const u32x4*>(ap);
    u32x4 ahi = *reinterpret_cast<const u32x4*>(ap + 8);
    // B-frag: row lm of lora_A, 16 contiguous K per lane group, same swizzle
    const f32x4* bp = reinterpret_cast<const f32x4*>(A + (size_t)lm * IN_F + k0 + hk * 16);
    f32x4 e0 = bp[0], o0 = bp[1], e1 = bp[2], o1 = bp[3];
    u32x4 blo = { packh(e0.x,o0.x), packh(e0.y,o0.y), packh(e0.z,o0.z), packh(e0.w,o0.w) };
    u32x4 bhi = { packh(e1.x,o1.x), packh(e1.y,o1.y), packh(e1.z,o1.z), packh(e1.w,o1.w) };
    acc = __builtin_amdgcn_wmma_f32_16x16x32_f16(false, mkfrag(alo, ahi),
                                                 false, mkfrag(blo, bhi),
                                                 (short)0, acc, false, false);
  }
  #pragma unroll
  for (int j = 0; j < 8; ++j) {
    const int m = m0 + wave * 16 + j + (hk << 3);
    T[(size_t)m * LR + lm] = acc[j];
  }
}

// ---------------------------------------------------------------------------
// Kernel 2: out = x @ dequant(qweight) + bias + SCALING * T @ lora_B^T
// 128x128 tile/block, 8 waves in 2(M) x 4(N), 64x32 per wave. A-frags read
// f16 x directly from global (L2-resident); W dequanted int4->f16 with packed
// magic-number math into a double-buffered LDS tile (one barrier per K-step).
// ---------------------------------------------------------------------------
__global__ void __launch_bounds__(256)
gptq_lora_gemm(const unsigned short* __restrict__ xbf,
               const int*   __restrict__ qweight,
               const int*   __restrict__ qzeros,
               const float* __restrict__ scales,
               const float* __restrict__ bias,
               const float* __restrict__ loraB,
               const float* __restrict__ T,
               float* __restrict__ out) {
  // 32 f16 + pad -> 40 elems (80 B) per column row; 16B-aligned, conflict-free
  // for ds_load_b128 fragment reads. Two buffers (ping-pong).
  __shared__ __align__(16) unsigned short wtile[2][128 * 40];  // [n][k] f16

  const int t    = threadIdx.x;
  const int n0   = blockIdx.x * 128;
  const int m0   = blockIdx.y * 128;
  const int wave = t >> 5, lane = t & 31;
  const int wm   = wave >> 2, wn = wave & 3;       // 2 x 4 wave grid
  const int lm   = lane & 15, hk = lane >> 4;
  // W-staging role: 2 threads per output column
  const int wcol = t & 127, wh = t >> 7;
  const int nIdx = n0 + wcol;

  v8f acc[4][2];
  #pragma unroll
  for (int r = 0; r < 4; ++r)
    #pragma unroll
    for (int c = 0; c < 2; ++c) acc[r][c] = (v8f){};

  // ---- dequant stager: tile kt -> wtile[bsel] -------------------------------
  // nibble v -> f16(1024+v) via (q>>4i)&0x000F000F | 0x64006400 (exact),
  // then (val - (1024+z)) * s in packed f16 (sub exact, one rounding).
  auto stage = [&](int kt, int bsel) {
    const int k0   = kt * 32;
    const int g    = k0 >> 7;                     // 32 | 128 -> uniform group
    const int qrow = (k0 >> 3) + 2 * wh;
    const unsigned q0 = (unsigned)qweight[(size_t)qrow * OUT_F + nIdx];
    const unsigned q1 = (unsigned)qweight[(size_t)(qrow + 1) * OUT_F + nIdx];
    const int z  = (qzeros[(size_t)g * (OUT_F / 8) + (nIdx >> 3)] >> ((nIdx & 7) * 4)) & 0xF;
    const float s = scales[(size_t)g * OUT_F + nIdx];
    const h2 kpk = __builtin_bit_cast(h2, 0x64006400u + (unsigned)z * 0x10001u);
    const h2 spk = __builtin_bit_cast(h2, packh(s, s));
    unsigned r[8];
    #pragma unroll
    for (int i = 0; i < 4; ++i) {
      unsigned b = (((q0 >> (4 * i)) & 0x000F000Fu) | 0x64006400u);
      h2 d = __builtin_bit_cast(h2, b) - kpk;          // (v - z), exact
      r[i] = __builtin_bit_cast(unsigned, d * spk);    // v_pk_mul_f16
    }
    #pragma unroll
    for (int i = 0; i < 4; ++i) {
      unsigned b = (((q1 >> (4 * i)) & 0x000F000Fu) | 0x64006400u);
      h2 d = __builtin_bit_cast(h2, b) - kpk;
      r[4 + i] = __builtin_bit_cast(unsigned, d * spk);
    }
    u32x4 p0 = { r[0], r[1], r[2], r[3] };
    u32x4 p1 = { r[4], r[5], r[6], r[7] };
    u32x4* dst = reinterpret_cast<u32x4*>(&wtile[bsel][wcol * 40 + wh * 16]);
    dst[0] = p0; dst[1] = p1;
  };

  stage(0, 0);
  __syncthreads();

  for (int kt = 0; kt < KSTEPS; ++kt) {
    const int b  = kt & 1;
    const int k0 = kt * 32;
    if (kt + 1 < KSTEPS) stage(kt + 1, b ^ 1);   // overlaps with WMMAs below

    v16h bfr[2];
    #pragma unroll
    for (int c = 0; c < 2; ++c) {
      const unsigned short* bp = &wtile[b][(wn * 32 + c * 16 + lm) * 40 + hk * 16];
      bfr[c] = mkfrag(*reinterpret_cast<const u32x4*>(bp),
                      *reinterpret_cast<const u32x4*>(bp + 8));
    }
    #pragma unroll
    for (int r = 0; r < 4; ++r) {
      const unsigned short* ap =
          xbf + (size_t)(m0 + wm * 64 + r * 16 + lm) * IN_F + k0 + hk * 16;
      v16h af = mkfrag(*reinterpret_cast<const u32x4*>(ap),
                       *reinterpret_cast<const u32x4*>(ap + 8));
      #pragma unroll
      for (int c = 0; c < 2; ++c)
        acc[r][c] = __builtin_amdgcn_wmma_f32_16x16x32_f16(false, af, false, bfr[c],
                                                           (short)0, acc[r][c], false, false);
    }
    __syncthreads();
  }

  // Epilogue: + bias + SCALING * <T[m], loraB[n]> (L2-resident, fp32)
  float bn[2];
  f32x4 B[2][4];
  #pragma unroll
  for (int c = 0; c < 2; ++c) {
    const int n = n0 + wn * 32 + c * 16 + lm;
    bn[c] = bias[n];
    const f32x4* Bp = reinterpret_cast<const f32x4*>(loraB + (size_t)n * LR);
    B[c][0] = Bp[0]; B[c][1] = Bp[1]; B[c][2] = Bp[2]; B[c][3] = Bp[3];
  }
  #pragma unroll
  for (int r = 0; r < 4; ++r) {
    #pragma unroll
    for (int j = 0; j < 8; ++j) {
      const int m = m0 + wm * 64 + r * 16 + j + (hk << 3);
      const f32x4* Tp = reinterpret_cast<const f32x4*>(T + (size_t)m * LR);
      f32x4 t0 = Tp[0], t1 = Tp[1], t2 = Tp[2], t3 = Tp[3];
      #pragma unroll
      for (int c = 0; c < 2; ++c) {
        const int n = n0 + wn * 32 + c * 16 + lm;
        float dot = t0.x*B[c][0].x + t0.y*B[c][0].y + t0.z*B[c][0].z + t0.w*B[c][0].w
                  + t1.x*B[c][1].x + t1.y*B[c][1].y + t1.z*B[c][1].z + t1.w*B[c][1].w
                  + t2.x*B[c][2].x + t2.y*B[c][2].y + t2.z*B[c][2].z + t2.w*B[c][2].w
                  + t3.x*B[c][3].x + t3.y*B[c][3].y + t3.z*B[c][3].z + t3.w*B[c][3].w;
        out[(size_t)m * OUT_F + n] = acc[r][c][j] + bn[c] + SCALING * dot;
      }
    }
  }
}

extern "C" void kernel_launch(void* const* d_in, const int* in_sizes, int n_in,
                              void* d_out, int out_size, void* d_ws, size_t ws_size,
                              hipStream_t stream) {
  const float* x       = (const float*)d_in[0];
  const int*   qweight = (const int*)  d_in[1];
  const int*   qzeros  = (const int*)  d_in[2];
  const float* scales  = (const float*)d_in[3];
  const float* bias    = (const float*)d_in[4];
  const float* lora_A  = (const float*)d_in[5];
  const float* lora_B  = (const float*)d_in[6];
  float* out = (float*)d_out;

  // workspace layout: [ xbf : 64 MB ][ T : 512 KB ]
  unsigned short* xbf = (unsigned short*)d_ws;
  float* T = (float*)((char*)d_ws + (size_t)TOKENS * IN_F * sizeof(unsigned short));

  x_to_f16_kernel<<<(TOKENS * (size_t)IN_F) / (256 * 8), 256, 0, stream>>>(x, xbf);
  lora_t_kernel<<<TOKENS / 128, 256, 0, stream>>>(xbf, lora_A, T);
  gptq_lora_gemm<<<dim3(OUT_F / 128, TOKENS / 128), 256, 0, stream>>>(
      xbf, qweight, qzeros, scales, bias, lora_B, T, out);
}
